// EIISRS_87866440942280
// MI455X (gfx1250) — matrix operand order repaired and checked by reading, not compiled, not run on hardware
//
#include <hip/hip_runtime.h>
#include <hip/hip_bf16.h>
#include <stdint.h>

#define BATCH  4096
#define DIM    64
#define NNBR   50
#define NLAYER 3

typedef __attribute__((ext_vector_type(16))) _Float16     v16h;
typedef __attribute__((ext_vector_type(8)))  _Float16     h8;
typedef __attribute__((ext_vector_type(8)))  float        v8f;
typedef __attribute__((ext_vector_type(4)))  unsigned int u32x4;
typedef __attribute__((ext_vector_type(8)))  int          i32x8;
typedef __attribute__((ext_vector_type(4)))  int          i32x4;

#define HAS_TDM (__has_builtin(__builtin_amdgcn_tensor_load_to_lds) && \
                 __has_builtin(__builtin_amdgcn_s_wait_tensorcnt))

// ---------------------------------------------------------------------------
// Kernel 1: fused gathers + cumulative-mask neighbor sums + 3 GNN layers.
// One 64-thread block (2 waves) per batch row; thread t owns channel t.
// S_p = sum_k nbr[k]*mask^p computed in ONE pass (mask is re-applied per layer
// in the reference, so layer l uses S_{l+1}/K). Layers stay fp32; final user
// embedding is written as f16 for the WMMA GEMM.
// ---------------------------------------------------------------------------
__global__ __launch_bounds__(64) void user_prep_kernel(
    const int*   __restrict__ user_ids,
    const int*   __restrict__ nbrs,
    const float* __restrict__ amask,
    const float* __restrict__ utab,
    const float* __restrict__ Ws,
    const float* __restrict__ bs,
    _Float16*    __restrict__ U2)
{
  const int b = blockIdx.x;
  const int t = threadIdx.x;
  __shared__ float xs[DIM];

  float u = utab[user_ids[b] * DIM + t];

  float s1 = 0.f, s2 = 0.f, s3 = 0.f;
  const int base = b * NNBR;
  for (int k = 0; k < NNBR; ++k) {
    const float m  = amask[base + k];          // broadcast (same addr all lanes)
    const int   nb = nbrs[base + k];
    const float v  = utab[nb * DIM + t];       // coalesced 256B row gather
    float vm = v * m;  s1 += vm;
    vm *= m;           s2 += vm;
    vm *= m;           s3 += vm;
  }
  const float inv = 1.0f / (float)NNBR;
  float s[NLAYER] = { s1 * inv, s2 * inv, s3 * inv };

  for (int l = 0; l < NLAYER; ++l) {
    __syncthreads();
    xs[t] = u + s[l];
    __syncthreads();
    float acc = bs[l * DIM + t];
    const float* wrow = &Ws[(l * DIM + t) * DIM];   // W row i: out = x @ W^T
    #pragma unroll 8
    for (int j = 0; j < DIM; ++j) acc = fmaf(xs[j], wrow[j], acc);
    u = fmaxf(acc, 0.0f);
  }
  U2[b * DIM + t] = (_Float16)u;
}

// ---------------------------------------------------------------------------
// Kernel 2: gather item embeddings, convert to f16, store K-major (transposed)
// so the GEMM's B-tile is a strided-2D tile the TDM can fetch directly.
// ---------------------------------------------------------------------------
__global__ __launch_bounds__(64) void item_prep_kernel(
    const int*   __restrict__ item_ids,
    const float* __restrict__ itab,
    _Float16*    __restrict__ ET)     // [DIM][BATCH]
{
  const int b = blockIdx.x;
  const int t = threadIdx.x;
  ET[t * BATCH + b] = (_Float16)itab[item_ids[b] * DIM + t];
}

// ---------------------------------------------------------------------------
// TDM descriptor builders (CDNA5 ISA ch.8, D# groups). 2-byte elements.
// ---------------------------------------------------------------------------
__device__ __forceinline__ u32x4 tdm_group0(const void* gptr, const void* ldsptr)
{
  const uint64_t ga = (uint64_t)(uintptr_t)gptr;     // 57-bit global tile addr
  const uint32_t la = (uint32_t)(uintptr_t)ldsptr;   // LDS byte offset (addr[31:0])
  u32x4 g;
  g.x = 1u;                                          // count=1, user descriptor
  g.y = la;                                          // lds_addr  (bits 63:32)
  g.z = (uint32_t)ga;                                // global_addr[31:0]
  g.w = (uint32_t)((ga >> 32) & 0x01FFFFFFu)         // global_addr[56:32]
      | (2u << 30);                                  // type=2 ("image")
  return g;
}

__device__ __forceinline__ i32x8 tdm_group1(uint32_t tensor_d0, uint32_t tensor_d1,
                                            uint32_t tile_d0,   uint32_t tile_d1,
                                            uint64_t d0_stride,
                                            uint32_t pad_interval, uint32_t pad_amount)
{
  // 256-bit little-endian bitfield, composed as 4 qwords.
  const uint64_t q0 = (1ull << 16)                           // data_size = 2 bytes
                    | (1ull << 20)                           // pad_enable
                    | ((uint64_t)pad_interval << 22)         // 2^(v+1) dwords
                    | ((uint64_t)pad_amount   << 25)         // v+1 dwords of pad
                    | ((uint64_t)(tensor_d0 & 0xFFFFu) << 48);  // dim0[15:0] @48
  const uint64_t q1 = ((uint64_t)tensor_d0 >> 16)            // dim0[31:16] @64
                    | ((uint64_t)tensor_d1 << 16)            // dim1 @80
                    | ((uint64_t)(tile_d0 & 0xFFFFu) << 48); // tile0 @112
  const uint64_t q2 = (uint64_t)(tile_d1 & 0xFFFFu)          // tile1 @128 (tile2=0)
                    | ((d0_stride & 0xFFFFFFFFull) << 32);   // stride0[31:0] @160
  const uint64_t q3 = (d0_stride >> 32) & 0xFFFFull;         // stride0[47:32] @192
  i32x8 g;
  g[0] = (int)(uint32_t)q0;  g[1] = (int)(uint32_t)(q0 >> 32);
  g[2] = (int)(uint32_t)q1;  g[3] = (int)(uint32_t)(q1 >> 32);
  g[4] = (int)(uint32_t)q2;  g[5] = (int)(uint32_t)(q2 >> 32);
  g[6] = (int)(uint32_t)q3;  g[7] = (int)(uint32_t)(q3 >> 32);
  return g;
}

// ---------------------------------------------------------------------------
// Kernel 3: scores[4096,4096] = U2[4096,64] @ E[4096,64]^T via
// v_wmma_f32_16x16x32_f16. 128x128 tile / 256-thread block (8 wave32).
// Tiles staged global->LDS by the Tensor Data Mover; the D# pad feature
// recreates the bank-conflict padding (A: 72-half rows, B: 136-half rows).
// Wave w owns rows [w*16, w*16+16) x 8 N-tiles -> 8 v8f accumulators,
// K=64 -> 2 WMMA k-steps per tile. HBM-store-bound (~2.9us floor @ 23.3TB/s).
// ---------------------------------------------------------------------------
#define TM        128
#define AS_STRIDE 72    // halves: 144B rows -> 16B-aligned b128, conflict-free
#define BS_STRIDE 136   // halves: 272B rows -> 16B-aligned b128

__global__ __launch_bounds__(256) void score_gemm_kernel(
    const _Float16* __restrict__ U2,   // [BATCH][DIM]  row-major
    const _Float16* __restrict__ ET,   // [DIM][BATCH]  k-major
    float*          __restrict__ C)    // [BATCH][BATCH]
{
  __shared__ _Float16 As[TM * AS_STRIDE];    // [m][k], 72-half stride
  __shared__ _Float16 Bs[DIM * BS_STRIDE];   // [k][n], 136-half stride

  const int bn0 = blockIdx.x * TM;
  const int bm0 = blockIdx.y * TM;
  const int tid = threadIdx.x;

#if HAS_TDM
  // --- TDM staging: wave 0 issues two tensor loads, waits TENSORcnt, then
  // --- releases the workgroup barrier so every wave sees valid LDS.
  if (tid < 32) {
    // A tile: 128 rows x 64 halves, row stride 64 halves in memory;
    // pad 4 dwords after every 32 dwords stored -> 72-half LDS rows.
    const u32x4 g0a = tdm_group0(U2 + (size_t)bm0 * DIM, &As[0]);
    const i32x8 g1a = tdm_group1(/*tensor_d0=*/DIM, /*tensor_d1=*/TM,
                                 /*tile_d0=*/DIM,   /*tile_d1=*/TM,
                                 /*d0_stride=*/DIM,
                                 /*pad_interval=*/4 /*32 dw*/, /*pad_amount=*/3 /*4 dw*/);
    // B tile: 64 rows x 128 halves, row stride 4096 halves in memory;
    // pad 4 dwords after every 64 dwords stored -> 136-half LDS rows.
    const u32x4 g0b = tdm_group0(ET + bn0, &Bs[0]);
    const i32x8 g1b = tdm_group1(/*tensor_d0=*/TM, /*tensor_d1=*/DIM,
                                 /*tile_d0=*/TM,   /*tile_d1=*/DIM,
                                 /*d0_stride=*/BATCH,
                                 /*pad_interval=*/5 /*64 dw*/, /*pad_amount=*/3 /*4 dw*/);
    const i32x4 z4 = {0, 0, 0, 0};               // groups 2/3 unused (2D tile)
    const i32x8 z8 = {0, 0, 0, 0, 0, 0, 0, 0};   // 6-arg form's extra group
    __builtin_amdgcn_tensor_load_to_lds(g0a, g1a, z4, z4, z8, /*cpol=*/0);
    __builtin_amdgcn_tensor_load_to_lds(g0b, g1b, z4, z4, z8, /*cpol=*/0);
    __builtin_amdgcn_s_wait_tensorcnt(0);
  }
  __syncthreads();
#else
  // --- Fallback staging: 1024 16-byte chunks per tile through VGPRs.
  #pragma unroll
  for (int it = 0; it < 4; ++it) {
    const int idx = tid + it * 256;            // 0..1023
    const int m  = idx >> 3;                   // A: 8 chunks per row
    const int kc = (idx & 7) << 3;
    *(h8*)&As[m * AS_STRIDE + kc] = *(const h8*)&U2[(bm0 + m) * DIM + kc];
    const int k  = idx >> 4;                   // B: 16 chunks per row
    const int nc = (idx & 15) << 3;
    *(h8*)&Bs[k * BS_STRIDE + nc] = *(const h8*)&ET[k * BATCH + bn0 + nc];
  }
  __syncthreads();
#endif

  const int w    = tid >> 5;     // wave id -> M-tile strip
  const int lane = tid & 31;
  const int m16  = lane & 15;
  const int h    = lane >> 4;

  v8f acc[8] = {};               // 8 N-tiles of 16x16 f32

  #pragma unroll
  for (int k0 = 0; k0 < DIM; k0 += 32) {
    // A fragment (ISA 16-bit A 16x32 layout): lane holds M=lane%16;
    // elements e<8 -> K = k0 + 8h + e ; e>=8 -> K = k0 + 8h + 16 + (e-8).
    const _Float16* arow = &As[(w * 16 + m16) * AS_STRIDE + k0 + 8 * h];
    const h8 a0 = *(const h8*)(arow);
    const h8 a1 = *(const h8*)(arow + 16);
    v16h a;
    #pragma unroll
    for (int e = 0; e < 8; ++e) { a[e] = a0[e]; a[e + 8] = a1[e]; }

    #pragma unroll
    for (int nt = 0; nt < 8; ++nt) {
      // B fragment (32x16): lane holds N = nt*16 + lane%16;
      // elements e<8 -> K = k0 + 8h + e ; e>=8 -> K = k0 + 8h + 16 + (e-8).
      const _Float16* bcol = &Bs[(k0 + 8 * h) * BS_STRIDE + nt * 16 + m16];
      v16h bf;
      #pragma unroll
      for (int e = 0; e < 8; ++e) { bf[e] = bcol[e * BS_STRIDE]; bf[e + 8] = bcol[(e + 16) * BS_STRIDE]; }

      acc[nt] = __builtin_amdgcn_wmma_f32_16x16x32_f16(
          /*neg_a=*/false, a, /*neg_b=*/false, bf,
          /*c_mod=*/(short)0, acc[nt],
          /*reuse_a=*/false, /*reuse_b=*/false);
    }
  }

  // C/D layout: VGPR r, lane -> (M = r + 8h, N = lane%16).
  #pragma unroll
  for (int nt = 0; nt < 8; ++nt) {
    const int col = bn0 + nt * 16 + m16;
    #pragma unroll
    for (int r = 0; r < 8; ++r) {
      C[(size_t)(bm0 + w * 16 + 8 * h + r) * BATCH + col] = acc[nt][r];
    }
  }
}

// ---------------------------------------------------------------------------
extern "C" void kernel_launch(void* const* d_in, const int* in_sizes, int n_in,
                              void* d_out, int out_size, void* d_ws, size_t ws_size,
                              hipStream_t stream) {
  const int*   user_ids = (const int*)  d_in[0];
  const int*   item_ids = (const int*)  d_in[1];
  const int*   nbrs     = (const int*)  d_in[2];
  const float* amask    = (const float*)d_in[3];
  const float* utab     = (const float*)d_in[4];
  const float* itab     = (const float*)d_in[5];
  const float* Ws       = (const float*)d_in[6];
  const float* bs       = (const float*)d_in[7];
  float* C = (float*)d_out;

  _Float16* U2 = (_Float16*)d_ws;                                      // 512 KB
  _Float16* ET = (_Float16*)((char*)d_ws + (size_t)BATCH * DIM * 2);   // 512 KB

  user_prep_kernel<<<BATCH, 64, 0, stream>>>(user_ids, nbrs, amask, utab, Ws, bs, U2);
  item_prep_kernel<<<BATCH, 64, 0, stream>>>(item_ids, itab, ET);

  dim3 grid(BATCH / TM, BATCH / TM);  // 32 x 32
  score_gemm_kernel<<<grid, 256, 0, stream>>>(U2, ET, C);
}
